// BIAN_73057393705164
// MI455X (gfx1250) — matrix operand order repaired and checked by reading, not compiled, not run on hardware
//
#include <hip/hip_runtime.h>
#include <hip/hip_bf16.h>

typedef __attribute__((ext_vector_type(16))) __bf16 v16bf;
typedef __attribute__((ext_vector_type(8)))  float  v8f;

union bfpk { __bf16 h[2]; unsigned int u; };

// ---------------------------------------------------------------- utilities
__global__ void fill_kernel(float* __restrict__ p, size_t n, float v) {
    size_t i = (size_t)blockIdx.x * blockDim.x + threadIdx.x;
    size_t stride = (size_t)gridDim.x * blockDim.x;
    for (; i < n; i += stride) p[i] = v;
}

__global__ void deg_cnt_kernel(const int* __restrict__ src, const int* __restrict__ dst,
                               float* __restrict__ deg, float* __restrict__ cnt, int E) {
    int e = blockIdx.x * blockDim.x + threadIdx.x;
    if (e >= E) return;
    atomicAdd(&deg[dst[e]], 1.0f);
    atomicAdd(&cnt[src[e]], 1.0f);
}

__global__ void dinv_kernel(float* __restrict__ deg_to_dinv, int N) {
    int v = blockIdx.x * blockDim.x + threadIdx.x;
    if (v >= N) return;
    // reference: deg includes +1 self loop; deg>0 always -> plain rsqrt
    deg_to_dinv[v] = rsqrtf(deg_to_dinv[v] + 1.0f);
}

// ------------------------------------------------- generic WMMA GEMM (Nout=64)
// C[M,64] = act(A[M,K] @ W[K,64] + bias), K multiple of 32, K <= 256.
// 128 threads = 4 waves; wave w computes rows [blk*64 + 16w, +16).
// W and A staged in LDS PRE-SWIZZLED into WMMA fragment order: each lane's
// fragment = one contiguous 32B run -> ds_load_b128 pairs.
__launch_bounds__(128)
__global__ void gemm_wmma_kernel(const float* __restrict__ A, const float* __restrict__ W,
                                 const float* __restrict__ bias, float* __restrict__ C,
                                 int M, int K, int do_relu) {
    // sW fragment layout: [(k>>5)*4 + (n>>4)][lane=((k>>4)&1)*16 + (n&15)][e=k&15]
    __shared__ __align__(32) __bf16 sW[256 * 64];
    // sA fragment layout per wave: [lane][e] (16 bf16 = 32B per lane)
    __shared__ __align__(32) __bf16 sAf[4][32 * 16];
    __shared__ float sBias[64];

    const int tid = threadIdx.x;
    // stage W: pairs along K (k even) -> adjacent fragment elems -> b32 stores
    for (int p = tid; p < (K >> 1) * 64; p += 128) {
        int k = (p >> 6) << 1, n = p & 63;
        bfpk pk;
        pk.h[0] = (__bf16)W[k * 64 + n];
        pk.h[1] = (__bf16)W[(k + 1) * 64 + n];
        int idx = (((k >> 5) * 4 + (n >> 4)) << 9)
                + ((((k >> 4) & 1) << 4) + (n & 15)) * 16
                + (k & 15);
        *(unsigned int*)&sW[idx] = pk.u;
    }
    if (tid < 64) sBias[tid] = bias ? bias[tid] : 0.0f;
    __syncthreads();

    const int warp   = tid >> 5;
    const int lane   = tid & 31;
    const int lanelo = lane & 15;
    const int half   = lane >> 4;
    const int strip  = blockIdx.x * 64 + warp * 16;

    // A staging: lane owns column pair c0=2*(lane&15), rows r = (lane>>4)*8 + i.
    // Fragment dest: lane' = r + ((c0>>4)&1)*16, elem e = c0&15
    // (pairs never straddle fragment boundaries -> one b32 store per float2)
    const int c0        = lanelo * 2;
    const int rbase     = half * 8;
    const int dLaneOff  = ((c0 >> 4) & 1) << 4;
    const int dElem     = c0 & 15;
    const bool fullTile = (strip + 16 <= M);

    v8f acc[4] = {v8f{}, v8f{}, v8f{}, v8f{}};

    for (int kb = 0; kb < K; kb += 32) {
        if (fullTile) {
#pragma unroll
            for (int i = 0; i < 8; ++i) {
                int r = rbase + i;
                float2 vv = *(const float2*)&A[(size_t)(strip + r) * K + kb + c0];
                bfpk pk; pk.h[0] = (__bf16)vv.x; pk.h[1] = (__bf16)vv.y;
                *(unsigned int*)&sAf[warp][(r + dLaneOff) * 16 + dElem] = pk.u;
            }
        } else {
#pragma unroll
            for (int i = 0; i < 8; ++i) {
                int r = rbase + i;
                int m = strip + r;
                float2 vv = {0.0f, 0.0f};
                if (m < M) vv = *(const float2*)&A[(size_t)m * K + kb + c0];
                bfpk pk; pk.h[0] = (__bf16)vv.x; pk.h[1] = (__bf16)vv.y;
                *(unsigned int*)&sAf[warp][(r + dLaneOff) * 16 + dElem] = pk.u;
            }
        }
        // wave-private region; LDS in-order per wave -> no __syncthreads
        v16bf afrag = *(const v16bf*)&sAf[warp][lane * 16];

        const int kgrp = (kb >> 5) * 4;
#pragma unroll
        for (int t = 0; t < 4; ++t) {
            v16bf bfrag = *(const v16bf*)&sW[((kgrp + t) << 9) + lane * 16];
            acc[t] = __builtin_amdgcn_wmma_f32_16x16x32_bf16(
                false, afrag, false, bfrag, (short)0, acc[t], false, false);
        }
    }

    // C/D layout: lanes 0-15 -> M=j, lanes 16-31 -> M=8+j; N = lane%16
#pragma unroll
    for (int t = 0; t < 4; ++t) {
        int n = t * 16 + lanelo;
        float bb = sBias[n];
#pragma unroll
        for (int j = 0; j < 8; ++j) {
            int m = strip + j + half * 8;
            if (m < M) {
                float val = acc[t][j] + bb;
                if (do_relu) val = fmaxf(val, 0.0f);
                C[(size_t)m * 64 + n] = val;
            }
        }
    }
}

// ------------------------------------------------------------- GCN edge pass
// wave per edge: agg[dst] += h[src] * dinv[src]*dinv[dst]
__global__ void gcn_edge_kernel(const float* __restrict__ h, const int* __restrict__ src,
                                const int* __restrict__ dst, const float* __restrict__ dinv,
                                float* __restrict__ agg, int E) {
    int e = blockIdx.x * (blockDim.x >> 5) + (threadIdx.x >> 5);
    int lane = threadIdx.x & 31;
    if (e >= E) return;
    int s = src[e], d = dst[e];
    float nm = dinv[s] * dinv[d];
    float2 hv = ((const float2*)(h + (size_t)s * 64))[lane];
    float* ap = agg + (size_t)d * 64 + lane * 2;
    atomicAdd(ap,     hv.x * nm);
    atomicAdd(ap + 1, hv.y * nm);
}

// out = relu(agg + h*dinv^2 (self loop) + bias[c] + extra)
__global__ void gcn_final_kernel(const float* __restrict__ agg, const float* __restrict__ h,
                                 const float* __restrict__ dinv, const float* __restrict__ bias,
                                 const float* __restrict__ extra, float* __restrict__ out, int N) {
    int i = blockIdx.x * blockDim.x + threadIdx.x;
    if (i >= N * 64) return;
    int v = i >> 6, c = i & 63;
    float di = dinv[v];
    float val = agg[i] + h[i] * di * di + bias[c];
    if (extra) val += extra[i];
    out[i] = fmaxf(val, 0.0f);
}

// ------------------------------------------------------------ fusion scatter
// nf[src][64:256] += [x2[dst], edge_enc[e], T2[ts[e]]]  (block 0 == cnt*x2 -> finalize)
__global__ void fusion_edge_kernel(const float* __restrict__ x2, const float* __restrict__ eenc,
                                   const float* __restrict__ T2, const int* __restrict__ src,
                                   const int* __restrict__ dst, const int* __restrict__ ts,
                                   float* __restrict__ nf, int E) {
    int e = blockIdx.x * (blockDim.x >> 5) + (threadIdx.x >> 5);
    int lane = threadIdx.x & 31;
    if (e >= E) return;
    int s = src[e], d = dst[e];
    int t = ts[e]; t = t < 0 ? 0 : (t > 999 ? 999 : t);
    float2 xd = ((const float2*)(x2   + (size_t)d * 64))[lane];
    float2 ev = ((const float2*)(eenc + (size_t)e * 64))[lane];
    float2 tv = ((const float2*)(T2   + (size_t)t * 64))[lane];
    float* row = nf + (size_t)s * 256 + lane * 2;
    atomicAdd(row + 64,  xd.x); atomicAdd(row + 65,  xd.y);
    atomicAdd(row + 128, ev.x); atomicAdd(row + 129, ev.y);
    atomicAdd(row + 192, tv.x); atomicAdd(row + 193, tv.y);
}

// per-node: divide by max(cnt,1), block0 = x2 (==cnt*x2/cnt), empty-mask fallback
__global__ void fusion_final_kernel(const float* __restrict__ x2, const float* __restrict__ x_enc,
                                    const float* __restrict__ cnt, float* __restrict__ nf, int N) {
    int v = blockIdx.x * (blockDim.x >> 5) + (threadIdx.x >> 5);
    int lane = threadIdx.x & 31;
    if (v >= N) return;
    float c  = cnt[v];
    float inv = 1.0f / fmaxf(c, 1.0f);
    float* row = nf + (size_t)v * 256;
    float2 x2v = ((const float2*)(x2 + (size_t)v * 64))[lane];
    float b0x = (c > 0.0f) ? x2v.x : 0.0f;
    float b0y = (c > 0.0f) ? x2v.y : 0.0f;
    float2 b1 = ((const float2*)(row + 64))[lane];  b1.x *= inv; b1.y *= inv;
    float2 b2 = ((const float2*)(row + 128))[lane]; b2.x *= inv; b2.y *= inv;
    float2 b3 = ((const float2*)(row + 192))[lane]; b3.x *= inv; b3.y *= inv;
    float sum = fabsf(b0x) + fabsf(b0y) + fabsf(b1.x) + fabsf(b1.y)
              + fabsf(b2.x) + fabsf(b2.y) + fabsf(b3.x) + fabsf(b3.y);
#pragma unroll
    for (int m = 16; m; m >>= 1) sum += __shfl_xor(sum, m, 32);
    if (sum < 1e-6f) {  // no-outgoing-edge fallback: tile x_enc to width 256
        float2 xe = ((const float2*)(x_enc + (size_t)v * 64))[lane];
        b0x = xe.x; b0y = xe.y; b1 = xe; b2 = xe; b3 = xe;
    }
    float2 b0 = {b0x, b0y};
    ((float2*)(row      ))[lane] = b0;
    ((float2*)(row + 64 ))[lane] = b1;
    ((float2*)(row + 128))[lane] = b2;
    ((float2*)(row + 192))[lane] = b3;
}

// --------------------------------------------------- tiny final head (64->2)
__global__ void head_kernel(const float* __restrict__ hid, const float* __restrict__ Wm2,
                            const float* __restrict__ bm2, float* __restrict__ out, int N) {
    int v = blockIdx.x * blockDim.x + threadIdx.x;
    if (v >= N) return;
    float a0 = bm2[0], a1 = bm2[1];
    const float* hp = hid + (size_t)v * 64;
#pragma unroll 8
    for (int c = 0; c < 64; ++c) {
        float hh = hp[c];
        a0 += hh * Wm2[c * 2];
        a1 += hh * Wm2[c * 2 + 1];
    }
    out[(size_t)v * 2]     = a0;
    out[(size_t)v * 2 + 1] = a1;
}

// ---------------------------------------------------------------------------
extern "C" void kernel_launch(void* const* d_in, const int* in_sizes, int n_in,
                              void* d_out, int out_size, void* d_ws, size_t ws_size,
                              hipStream_t stream) {
    const float* x    = (const float*)d_in[0];
    const int*   eidx = (const int*)  d_in[1];
    const float* ea   = (const float*)d_in[2];
    const int*   ts   = (const int*)  d_in[3];
    const float* Wn   = (const float*)d_in[4];
    const float* bn   = (const float*)d_in[5];
    const float* We   = (const float*)d_in[6];
    const float* be   = (const float*)d_in[7];
    const float* Temb = (const float*)d_in[8];
    const float* Wt   = (const float*)d_in[9];
    const float* bt   = (const float*)d_in[10];
    const float* Wc1  = (const float*)d_in[11];
    const float* bc1  = (const float*)d_in[12];
    const float* Wc2  = (const float*)d_in[13];
    const float* bc2  = (const float*)d_in[14];
    const float* Wr   = (const float*)d_in[15];
    const float* br   = (const float*)d_in[16];
    const float* Wm1  = (const float*)d_in[17];
    const float* bm1  = (const float*)d_in[18];
    const float* Wm2  = (const float*)d_in[19];
    const float* bm2  = (const float*)d_in[20];

    const int N = in_sizes[0] / 128;
    const int E = in_sizes[1] / 2;
    const int* src = eidx;
    const int* dst = eidx + E;

    float* ws = (float*)d_ws;
    size_t off = 0;
    float* x_enc = ws + off; off += (size_t)N * 64;
    float* htmp  = ws + off; off += (size_t)N * 64;   // h1 -> h2 -> hid
    float* agg   = ws + off; off += (size_t)N * 64;
    float* xcur  = ws + off; off += (size_t)N * 64;   // x1 -> x2
    float* resid = ws + off; off += (size_t)N * 64;
    float* eenc  = ws + off; off += (size_t)E * 64;
    float* T2    = ws + off; off += (size_t)1024 * 64;
    float* nf    = ws + off; off += (size_t)N * 256;
    float* dinv  = ws + off; off += (size_t)N;        // deg -> dinv in place
    float* cnt   = ws + off; off += (size_t)N;

    const int gN  = (N + 63) / 64;           // gemm blocks over node rows
    const int gE  = (E + 63) / 64;           // gemm blocks over edge rows
    const int eBl = (E + 7) / 8;             // 8 edges per 256-thread block
    const int nBl = (N + 7) / 8;             // 8 nodes per 256-thread block

    // degrees / out-counts
    fill_kernel<<<2048, 256, 0, stream>>>(dinv, (size_t)N, 0.0f);
    fill_kernel<<<2048, 256, 0, stream>>>(cnt,  (size_t)N, 0.0f);
    deg_cnt_kernel<<<(E + 255) / 256, 256, 0, stream>>>(src, dst, dinv, cnt, E);
    dinv_kernel<<<(N + 255) / 256, 256, 0, stream>>>(dinv, N);

    // encoders (WMMA)
    gemm_wmma_kernel<<<gN, 128, 0, stream>>>(x, Wn, bn, x_enc, N, 128, 1);
    gemm_wmma_kernel<<<gE, 128, 0, stream>>>(ea, We, be, eenc, E, 32, 1);
    gemm_wmma_kernel<<<(1000 + 63) / 64, 128, 0, stream>>>(Temb, Wt, bt, T2, 1000, 64, 1);

    // GCN layer 1
    gemm_wmma_kernel<<<gN, 128, 0, stream>>>(x_enc, Wc1, nullptr, htmp, N, 64, 0);
    fill_kernel<<<4096, 256, 0, stream>>>(agg, (size_t)N * 64, 0.0f);
    gcn_edge_kernel<<<eBl, 256, 0, stream>>>(htmp, src, dst, dinv, agg, E);
    gcn_final_kernel<<<((size_t)N * 64 + 255) / 256, 256, 0, stream>>>(
        agg, htmp, dinv, bc1, nullptr, xcur, N);

    // GCN layer 2 + residual
    gemm_wmma_kernel<<<gN, 128, 0, stream>>>(xcur, Wc2, nullptr, htmp, N, 64, 0);
    gemm_wmma_kernel<<<gN, 128, 0, stream>>>(x_enc, Wr, br, resid, N, 64, 0);
    fill_kernel<<<4096, 256, 0, stream>>>(agg, (size_t)N * 64, 0.0f);
    gcn_edge_kernel<<<eBl, 256, 0, stream>>>(htmp, src, dst, dinv, agg, E);
    gcn_final_kernel<<<((size_t)N * 64 + 255) / 256, 256, 0, stream>>>(
        agg, htmp, dinv, bc2, resid, xcur, N);   // xcur now = x2

    // fusion
    fill_kernel<<<8192, 256, 0, stream>>>(nf, (size_t)N * 256, 0.0f);
    fusion_edge_kernel<<<eBl, 256, 0, stream>>>(xcur, eenc, T2, src, dst, ts, nf, E);
    fusion_final_kernel<<<nBl, 256, 0, stream>>>(xcur, x_enc, cnt, nf, N);

    // classifier
    gemm_wmma_kernel<<<gN, 128, 0, stream>>>(nf, Wm1, bm1, htmp, N, 256, 1);  // htmp = hid
    head_kernel<<<(N + 255) / 256, 256, 0, stream>>>(htmp, Wm2, bm2, (float*)d_out, N);

    (void)n_in; (void)out_size; (void)ws_size; (void)bt; (void)bn;
}
